// mp_gnn_39943195852849
// MI455X (gfx1250) — matrix operand order repaired and checked by reading, not compiled, not run on hardware
//
#include <hip/hip_runtime.h>

// ---------------- problem constants ----------------
#define N_NODES 100000
#define N_EDGES 1600000
#define C 64
#define KE 224     // padded edge-feature K (3C+3=195 -> 224 = 7*32)
#define KES 232    // LDS A-tile row stride in halves (mult of 8 -> 16B rows)
#define KN 128     // node-MLP K (2C)
#define KNS 136    // LDS A-tile row stride for node kernel
#define EAS 72     // LDS intermediate (16x64) row stride in halves

typedef _Float16 half_t;
typedef _Float16 v16h __attribute__((ext_vector_type(16)));
typedef _Float16 v8h  __attribute__((ext_vector_type(8)));
typedef float    v8f  __attribute__((ext_vector_type(8)));

union V16HU { v16h v; uint4 q[2]; };

__device__ __forceinline__ v8f wmma_f16(v16h a, v16h b, v8f c) {
  // D = A(16x32 f16) * B(32x16 f16) + C(16x16 f32)
  return __builtin_amdgcn_wmma_f32_16x16x32_f16(
      /*neg_a=*/false, a, /*neg_b=*/false, b,
      /*c_mod=*/(short)0, c, /*reuse_a=*/false, /*reuse_b=*/false);
}

// A-fragment: row-major 16xK tile (stride in halves, rows 16B-aligned).
// 16-bit A layout: lane<16 -> M=lane, K = [kb..kb+8) and [kb+16..kb+24)
//                  lane>=16 -> M=lane-16, K = [kb+8..kb+16) and [kb+24..kb+32)
__device__ __forceinline__ v16h load_a_frag(const half_t* tile, int stride, int kb, int lane) {
  int m    = lane & 15;
  int koff = (lane >> 4) << 3;
  const half_t* base = tile + m * stride + kb + koff;
  V16HU u;
  u.q[0] = *(const uint4*)(base);
  u.q[1] = *(const uint4*)(base + 16);
  return u.v;
}

// B-fragment from global f16 weights stored transposed: Wt[n][k], row stride kpad
// (mult of 16 halves). 16-bit B layout: lane n=lane&15 holds column (ncol+n),
// lanes 0-15 -> K=[kb..kb+8),[kb+16..kb+24); lanes 16-31 -> K=[kb+8..kb+16),[kb+24..kb+32).
__device__ __forceinline__ v16h load_b_frag(const half_t* __restrict__ wt, int kpad,
                                            int ncol, int kb, int lane) {
  int n = lane & 15;
  int k = kb + ((lane & 16) ? 8 : 0);
  const half_t* base = wt + (long)(ncol + n) * kpad + k;
  V16HU u;
  u.q[0] = *(const uint4*)(base);
  u.q[1] = *(const uint4*)(base + 16);
  return u.v;
}

__device__ __forceinline__ float elu_f(float x) {
  return x > 0.0f ? x : (__expf(x) - 1.0f);
}

// Device pass (probe-confirmed in round 2): builtin exists for gfx1250 and
// lowers to native global_atomic_add_f32. Host semantic pass takes the stub.
__device__ __forceinline__ void atomAddF32(float* p, float v) {
#if defined(__HIP_DEVICE_COMPILE__) && __has_builtin(__builtin_amdgcn_global_atomic_fadd_f32)
  typedef __attribute__((address_space(1))) float GF;
  __builtin_amdgcn_global_atomic_fadd_f32((GF*)(__UINTPTR_TYPE__)p, v);
#else
  atomicAdd(p, v);   // host-pass stub only; never executed on device
#endif
}

// ---------------- weight prep: fp32 [K][Cout] -> f16 transposed+padded [Cout][Kpad]
__global__ void prep_weights_kernel(const float* __restrict__ W, half_t* __restrict__ out,
                                    int K, int Kpad, int Cout) {
  int idx = blockIdx.x * blockDim.x + threadIdx.x;
  if (idx >= Cout * Kpad) return;
  int n = idx / Kpad;
  int k = idx - n * Kpad;
  out[(long)n * Kpad + k] = (k < K) ? (half_t)W[(long)k * Cout + n] : (half_t)0.0f;
}

// ---------------- encoder: h = elu(x@W0+b0)@W1+b1
// Stage 1 (K=3) scalar per lane; stage 2 (64x64) via WMMA. One 16-node tile per wave.
__global__ __launch_bounds__(128) void encoder_kernel(
    const float* __restrict__ x,
    const float* __restrict__ W0, const float* __restrict__ b0,
    const half_t* __restrict__ W1t, const float* __restrict__ b1,
    float* __restrict__ h, half_t* __restrict__ h16) {
  __shared__ half_t Ttile[4][16 * EAS];
  int wave = threadIdx.x >> 5;
  int lane = threadIdx.x & 31;
  long tile = (long)blockIdx.x * 4 + wave;
  if (tile >= N_NODES / 16) return;     // wave-uniform
  long n0 = tile * 16;
  half_t* T = Ttile[wave];

  // stage 1: t = elu(x@W0+b0), 16 nodes x 64 ch; lane handles 32 ch of one node
  {
    int m  = lane & 15;
    int os = (lane >> 4) * 32;
    long node = n0 + m;
    float x0 = x[node * 3 + 0], x1 = x[node * 3 + 1], x2 = x[node * 3 + 2];
#pragma unroll
    for (int j = 0; j < 32; ++j) {
      int o = os + j;
      float v = elu_f(x0 * W0[o] + x1 * W0[C + o] + x2 * W0[2 * C + o] + b0[o]);
      T[m * EAS + o] = (half_t)v;
    }
  }
  asm volatile("" ::: "memory");

  // stage 2: (16 x 64) @ (64 x 64) WMMA
  v8f acc[4] = {};
#pragma unroll
  for (int c = 0; c < 2; ++c) {
    v16h a = load_a_frag(T, EAS, c * 32, lane);
#pragma unroll
    for (int t = 0; t < 4; ++t) {
      v16h b = load_b_frag(W1t, C, t * 16, c * 32, lane);
      acc[t] = wmma_f16(a, b, acc[t]);
    }
  }
  int n16 = lane & 15;
  int mo  = (lane >> 4) * 8;
#pragma unroll
  for (int r = 0; r < 8; ++r) {
    long node = n0 + mo + r;
#pragma unroll
    for (int t = 0; t < 4; ++t) {
      float v = acc[t][r] + b1[t * 16 + n16];
      h[node * C + t * 16 + n16]   = v;
      h16[node * C + t * 16 + n16] = (half_t)v;
    }
  }
}

// ---------------- fused edge MLP + scatter-add (one 16-edge tile per wave)
__global__ __launch_bounds__(128) void edge_mlp_kernel(
    const half_t* __restrict__ h16, const float* __restrict__ pos,
    const int* __restrict__ srcIdx, const int* __restrict__ dstIdx,
    const half_t* __restrict__ eW0t, const float* __restrict__ eb0,
    const half_t* __restrict__ eW1t, const float* __restrict__ eb1,
    float* __restrict__ agg) {
  __shared__ half_t Atile[4][16 * KES];
  __shared__ half_t Etile[4][16 * EAS];
  int wave = threadIdx.x >> 5;
  int lane = threadIdx.x & 31;
  long tile = (long)blockIdx.x * 4 + wave;
  if (tile >= N_EDGES / 16) return;   // wave-uniform
  long e0 = tile * 16;
  half_t* A  = Atile[wave];
  half_t* EA = Etile[wave];

  // ---- stage A = [h[src] | h[dst] | h[src]-h[dst] | pos_diff | 0pad], 16 x 224 f16
  {
    int e  = lane & 15;
    int hs = lane >> 4;                 // which half of the 64 channels
    int s = srcIdx[e0 + e];
    int d = dstIdx[e0 + e];
    const v8h* ps = (const v8h*)(h16 + (long)s * C + hs * 32);
    const v8h* pd = (const v8h*)(h16 + (long)d * C + hs * 32);
    v8h* An = (v8h*)(A + e * KES + hs * 32);
    v8h* Ao = (v8h*)(A + e * KES + C + hs * 32);
    v8h* Ad = (v8h*)(A + e * KES + 2 * C + hs * 32);
#pragma unroll
    for (int j = 0; j < 4; ++j) {
      v8h an = ps[j], ao = pd[j];
      An[j] = an; Ao[j] = ao; Ad[j] = an - ao;
    }
    if (hs == 0) {
      A[e * KES + 192] = (half_t)(pos[(long)s * 3 + 0] - pos[(long)d * 3 + 0]);
      A[e * KES + 193] = (half_t)(pos[(long)s * 3 + 1] - pos[(long)d * 3 + 1]);
      A[e * KES + 194] = (half_t)(pos[(long)s * 3 + 2] - pos[(long)d * 3 + 2]);
#pragma unroll
      for (int k = 195; k < 208; ++k) A[e * KES + k] = (half_t)0.0f;
    } else {
#pragma unroll
      for (int k = 208; k < KE; ++k) A[e * KES + k] = (half_t)0.0f;
    }
  }
  asm volatile("" ::: "memory");   // same-wave LDS is in-order; just stop reordering

  // ---- GEMM1: (16 x 224) @ (224 x 64) -> f32, 7 k-chunks x 4 n-tiles
  v8f acc[4] = {};
#pragma unroll
  for (int c = 0; c < 7; ++c) {
    v16h a = load_a_frag(A, KES, c * 32, lane);
#pragma unroll
    for (int t = 0; t < 4; ++t) {
      v16h b = load_b_frag(eW0t, KE, t * 16, c * 32, lane);
      acc[t] = wmma_f16(a, b, acc[t]);
    }
  }

  // ---- bias + ELU -> f16 intermediate tile in LDS
  int n16 = lane & 15;
  int mo  = (lane >> 4) * 8;          // D-layout: lanes>=16 hold rows M+8
#pragma unroll
  for (int t = 0; t < 4; ++t) {
    float bias = eb0[t * 16 + n16];
#pragma unroll
    for (int r = 0; r < 8; ++r)
      EA[(mo + r) * EAS + t * 16 + n16] = (half_t)elu_f(acc[t][r] + bias);
  }
  asm volatile("" ::: "memory");

  // ---- GEMM2: (16 x 64) @ (64 x 64)
  v8f a2[4] = {};
#pragma unroll
  for (int c = 0; c < 2; ++c) {
    v16h a = load_a_frag(EA, EAS, c * 32, lane);
#pragma unroll
    for (int t = 0; t < 4; ++t) {
      v16h b = load_b_frag(eW1t, C, t * 16, c * 32, lane);
      a2[t] = wmma_f16(a, b, a2[t]);
    }
  }

  // ---- bias + scatter-add to agg[dst]
#pragma unroll
  for (int r = 0; r < 8; ++r) {
    int m = mo + r;
    int d = dstIdx[e0 + m];
#pragma unroll
    for (int t = 0; t < 4; ++t)
      atomAddF32(agg + (long)d * C + t * 16 + n16, a2[t][r] + eb1[t * 16 + n16]);
  }
}

// ---------------- fused node MLP + residual (one 16-node tile per wave)
__global__ __launch_bounds__(128) void node_mlp_kernel(
    float* __restrict__ h, half_t* __restrict__ h16,
    const float* __restrict__ agg,
    const half_t* __restrict__ nW0t, const float* __restrict__ nb0,
    const half_t* __restrict__ nW1t, const float* __restrict__ nb1) {
  __shared__ half_t Atile[4][16 * KNS];
  __shared__ half_t Etile[4][16 * EAS];
  int wave = threadIdx.x >> 5;
  int lane = threadIdx.x & 31;
  long tile = (long)blockIdx.x * 4 + wave;
  if (tile >= N_NODES / 16) return;   // wave-uniform (N % 16 == 0)
  long n0 = tile * 16;
  half_t* A  = Atile[wave];
  half_t* EA = Etile[wave];

  // ---- stage A = [h | agg] 16 x 128 f16
  {
    int m  = lane & 15;
    int hs = lane >> 4;
    long node = n0 + m;
    const v8h* ph = (const v8h*)(h16 + node * C + hs * 32);
    v8h* Ah = (v8h*)(A + m * KNS + hs * 32);
#pragma unroll
    for (int j = 0; j < 4; ++j) Ah[j] = ph[j];
    const float* pa = agg + node * C + hs * 32;
    half_t* Aa = A + m * KNS + C + hs * 32;
#pragma unroll
    for (int j = 0; j < 8; ++j) {
      float4 f = ((const float4*)pa)[j];
      half_t* q = Aa + j * 4;
      q[0] = (half_t)f.x; q[1] = (half_t)f.y; q[2] = (half_t)f.z; q[3] = (half_t)f.w;
    }
  }
  asm volatile("" ::: "memory");

  // ---- GEMM1: (16 x 128) @ (128 x 64)
  v8f acc[4] = {};
#pragma unroll
  for (int c = 0; c < 4; ++c) {
    v16h a = load_a_frag(A, KNS, c * 32, lane);
#pragma unroll
    for (int t = 0; t < 4; ++t) {
      v16h b = load_b_frag(nW0t, KN, t * 16, c * 32, lane);
      acc[t] = wmma_f16(a, b, acc[t]);
    }
  }

  int n16 = lane & 15;
  int mo  = (lane >> 4) * 8;
#pragma unroll
  for (int t = 0; t < 4; ++t) {
    float bias = nb0[t * 16 + n16];
#pragma unroll
    for (int r = 0; r < 8; ++r)
      EA[(mo + r) * EAS + t * 16 + n16] = (half_t)elu_f(acc[t][r] + bias);
  }
  asm volatile("" ::: "memory");

  // ---- GEMM2: (16 x 64) @ (64 x 64)
  v8f a2[4] = {};
#pragma unroll
  for (int c = 0; c < 2; ++c) {
    v16h a = load_a_frag(EA, EAS, c * 32, lane);
#pragma unroll
    for (int t = 0; t < 4; ++t) {
      v16h b = load_b_frag(nW1t, C, t * 16, c * 32, lane);
      a2[t] = wmma_f16(a, b, a2[t]);
    }
  }

  // ---- bias + residual; refresh f32 and f16 copies of h
#pragma unroll
  for (int r = 0; r < 8; ++r) {
    long node = n0 + mo + r;
#pragma unroll
    for (int t = 0; t < 4; ++t) {
      float v  = a2[t][r] + nb1[t * 16 + n16];
      float hn = h[node * C + t * 16 + n16] + v;
      h[node * C + t * 16 + n16]   = hn;
      h16[node * C + t * 16 + n16] = (half_t)hn;
    }
  }
}

// ---------------- decoder: out = elu(h@W0+b0)@W1+b1   (64 -> 64 -> 3)
// Stage 1 (64x64) via WMMA, A-fragments straight from global h16; stage 2 (N=3)
// scalar with cross-lane reduction.
__global__ __launch_bounds__(128) void decoder_kernel(
    const half_t* __restrict__ h16,
    const half_t* __restrict__ W0t, const float* __restrict__ b0,
    const float* __restrict__ W1, const float* __restrict__ b1,
    float* __restrict__ out) {
  __shared__ half_t Etile[4][16 * EAS];
  int wave = threadIdx.x >> 5;
  int lane = threadIdx.x & 31;
  long tile = (long)blockIdx.x * 4 + wave;
  if (tile >= N_NODES / 16) return;    // wave-uniform
  long n0 = tile * 16;
  half_t* EA = Etile[wave];

  // ---- stage 1: (16 x 64) @ (64 x 64) WMMA, A direct from global h16
  v8f acc[4] = {};
#pragma unroll
  for (int c = 0; c < 2; ++c) {
    int m    = lane & 15;
    int koff = (lane >> 4) << 3;
    const half_t* base = h16 + (n0 + m) * C + c * 32 + koff;
    V16HU u;
    u.q[0] = *(const uint4*)(base);
    u.q[1] = *(const uint4*)(base + 16);
#pragma unroll
    for (int t = 0; t < 4; ++t) {
      v16h b = load_b_frag(W0t, C, t * 16, c * 32, lane);
      acc[t] = wmma_f16(u.v, b, acc[t]);
    }
  }
  int n16 = lane & 15;
  int mo  = (lane >> 4) * 8;
#pragma unroll
  for (int t = 0; t < 4; ++t) {
    float bias = b0[t * 16 + n16];
#pragma unroll
    for (int r = 0; r < 8; ++r)
      EA[(mo + r) * EAS + t * 16 + n16] = (half_t)elu_f(acc[t][r] + bias);
  }
  asm volatile("" ::: "memory");

  // ---- stage 2: 64 -> 3 projection; lane pair (m, m+16) splits K, shfl-combine
  {
    int m  = lane & 15;
    int hs = lane >> 4;
    float s0 = 0.f, s1 = 0.f, s2 = 0.f;
    const v8h* pe = (const v8h*)(EA + m * EAS + hs * 32);
#pragma unroll
    for (int j = 0; j < 4; ++j) {
      v8h tv = pe[j];
#pragma unroll
      for (int k = 0; k < 8; ++k) {
        int kk = hs * 32 + j * 8 + k;
        float tvf = (float)tv[k];
        s0 += tvf * W1[kk * 3 + 0];
        s1 += tvf * W1[kk * 3 + 1];
        s2 += tvf * W1[kk * 3 + 2];
      }
    }
    s0 += __shfl_xor(s0, 16, 32);
    s1 += __shfl_xor(s1, 16, 32);
    s2 += __shfl_xor(s2, 16, 32);
    if (hs == 0) {
      long node = n0 + m;
      out[node * 3 + 0] = s0 + b1[0];
      out[node * 3 + 1] = s1 + b1[1];
      out[node * 3 + 2] = s2 + b1[2];
    }
  }
}

// ---------------- host-side orchestration ----------------
extern "C" void kernel_launch(void* const* d_in, const int* in_sizes, int n_in,
                              void* d_out, int out_size, void* d_ws, size_t ws_size,
                              hipStream_t stream) {
  (void)in_sizes; (void)n_in; (void)out_size; (void)ws_size;
  const float* x      = (const float*)d_in[0];
  const float* pos    = (const float*)d_in[1];
  const int*   ei     = (const int*)d_in[2];          // [2][E] (int32 on device)
  const int*   srcI   = ei;
  const int*   dstI   = ei + N_EDGES;
  const float* encW0  = (const float*)d_in[3];
  const float* encB0  = (const float*)d_in[4];
  const float* encW1  = (const float*)d_in[5];
  const float* encB1  = (const float*)d_in[6];
  const float* decW0  = (const float*)d_in[7];
  const float* decB0  = (const float*)d_in[8];
  const float* decW1  = (const float*)d_in[9];
  const float* decB1  = (const float*)d_in[10];
  const float* mpEW0  = (const float*)d_in[11];       // [2][195][64]
  const float* mpEB0  = (const float*)d_in[12];       // [2][64]
  const float* mpEW1  = (const float*)d_in[13];       // [2][64][64]
  const float* mpEB1  = (const float*)d_in[14];
  const float* mpNW0  = (const float*)d_in[15];       // [2][128][64]
  const float* mpNB0  = (const float*)d_in[16];
  const float* mpNW1  = (const float*)d_in[17];       // [2][64][64]
  const float* mpNB1  = (const float*)d_in[18];
  float* out = (float*)d_out;

  // workspace carve-out (256B aligned)
  size_t off = 0;
  char* ws = (char*)d_ws;
  auto carve = [&](size_t bytes) -> void* {
    void* p = ws + off;
    off += (bytes + 255) & ~(size_t)255;
    return p;
  };
  float*  h     = (float*) carve((size_t)N_NODES * C * sizeof(float));
  half_t* h16   = (half_t*)carve((size_t)N_NODES * C * sizeof(half_t));
  float*  agg   = (float*) carve((size_t)N_NODES * C * sizeof(float));
  half_t* eW0t  = (half_t*)carve((size_t)2 * C * KE * sizeof(half_t));
  half_t* eW1t  = (half_t*)carve((size_t)2 * C * C  * sizeof(half_t));
  half_t* nW0t  = (half_t*)carve((size_t)2 * C * KN * sizeof(half_t));
  half_t* nW1t  = (half_t*)carve((size_t)2 * C * C  * sizeof(half_t));
  half_t* encW1t= (half_t*)carve((size_t)C * C * sizeof(half_t));
  half_t* decW0t= (half_t*)carve((size_t)C * C * sizeof(half_t));

  // convert / transpose / pad all WMMA-path weights to f16
  for (int i = 0; i < 2; ++i) {
    prep_weights_kernel<<<(C * KE + 255) / 256, 256, 0, stream>>>(
        mpEW0 + (size_t)i * 195 * C, eW0t + (size_t)i * C * KE, 195, KE, C);
    prep_weights_kernel<<<(C * C + 255) / 256, 256, 0, stream>>>(
        mpEW1 + (size_t)i * C * C, eW1t + (size_t)i * C * C, C, C, C);
    prep_weights_kernel<<<(C * KN + 255) / 256, 256, 0, stream>>>(
        mpNW0 + (size_t)i * KN * C, nW0t + (size_t)i * C * KN, KN, KN, C);
    prep_weights_kernel<<<(C * C + 255) / 256, 256, 0, stream>>>(
        mpNW1 + (size_t)i * C * C, nW1t + (size_t)i * C * C, C, C, C);
  }
  prep_weights_kernel<<<(C * C + 255) / 256, 256, 0, stream>>>(encW1, encW1t, C, C, C);
  prep_weights_kernel<<<(C * C + 255) / 256, 256, 0, stream>>>(decW0, decW0t, C, C, C);

  const int edgeTiles = N_EDGES / 16;   // 100000, divides evenly
  const int nodeTiles = N_NODES / 16;   // 6250
  const int nodeGrid  = (nodeTiles + 3) / 4;

  encoder_kernel<<<nodeGrid, 128, 0, stream>>>(
      x, encW0, encB0, encW1t, encB1, h, h16);

  for (int i = 0; i < 2; ++i) {
    (void)hipMemsetAsync(agg, 0, (size_t)N_NODES * C * sizeof(float), stream);
    edge_mlp_kernel<<<edgeTiles / 4, 128, 0, stream>>>(
        h16, pos, srcI, dstI,
        eW0t + (size_t)i * C * KE, mpEB0 + (size_t)i * C,
        eW1t + (size_t)i * C * C,  mpEB1 + (size_t)i * C, agg);
    node_mlp_kernel<<<nodeGrid, 128, 0, stream>>>(
        h, h16, agg,
        nW0t + (size_t)i * C * KN, mpNB0 + (size_t)i * C,
        nW1t + (size_t)i * C * C,  mpNB1 + (size_t)i * C);
  }

  decoder_kernel<<<nodeGrid, 128, 0, stream>>>(
      h16, decW0t, decB0, decW1, decB1, out);
}
